// EncoderLayer_21629455303352
// MI455X (gfx1250) — compile-verified, matrix operand order checked by reading, not run on hardware
//
#include <hip/hip_runtime.h>
#include <math.h>

// ---------------------------------------------------------------------------
// MI455X / gfx1250: wave32, v_wmma_f32_16x16x32_bf16 (fp32 accumulate).
// - LDS tiles are stored in WMMA *fragment order*: every fragment load is
//   2 x ds_load_b128, zero packing VALU in the inner loop.
// - GEMM B-tiles are fetched by the Tensor Data Mover (tensor_load_to_lds,
//   TENSORcnt fenced), overlapping DMA with WMMA; A-tiles use clause-grouped
//   global_load_b128 + global_prefetch.
// - fp32 -> bf16 uses native converts (<2 x __bf16> pack), not integer math.
// Reference's V projection is dead code -> skipped.
// ---------------------------------------------------------------------------

typedef __attribute__((ext_vector_type(16))) __bf16        v16bf;
typedef __attribute__((ext_vector_type(2)))  __bf16        v2bf;
typedef __attribute__((ext_vector_type(8)))  float         v8f;
typedef __attribute__((ext_vector_type(8)))  unsigned int  v8u;
typedef __attribute__((ext_vector_type(4)))  unsigned int  u32x4;
typedef __attribute__((ext_vector_type(8)))  int           i32x8;
typedef __attribute__((ext_vector_type(4)))  int           i32x4;

__device__ __forceinline__ unsigned int pack2(float a, float b) {
    v2bf t;
    t[0] = (__bf16)a;                          // native cvt, RTNE
    t[1] = (__bf16)b;
    return __builtin_bit_cast(unsigned int, t);
}
// 16-bit fragment permutation: k (0..31) -> (hi, vgpr v). Pair (k,k+1), k even,
// lives in uint slot  (row + hi*16)*8 + v  of a 256-uint fragment image.
__device__ __forceinline__ int khi(int k) { return (k >> 3) & 1; }
__device__ __forceinline__ int kv (int k) { return ((k & 16) >> 2) | ((k >> 1) & 3); }

__device__ __forceinline__ v16bf load_frag(const unsigned int* base, int lane) {
    const uint4* p = reinterpret_cast<const uint4*>(base + lane * 8);
    uint4 a = p[0], b = p[1];                  // 2 x ds_load_b128
    v8u u = {a.x, a.y, a.z, a.w, b.x, b.y, b.z, b.w};
    return __builtin_bit_cast(v16bf, u);
}

// ---------------------------------------------------------------------------
// Tensor Data Mover: 2D fp32 tile (32 rows x 128 cols, row stride N elements)
// from global B[k0..k0+31][col0..col0+127] into LDS at lds_byte (row-major).
// D# per CDNA5 ISA (group0/group1; groups 2-3 zero = 2D).
// 6-arg builtin form (clang-23 / therock headers): extra zero int32x8 group.
// ---------------------------------------------------------------------------
__device__ __forceinline__ void tdm_load_b_tile(const float* gptr, unsigned lds_byte,
                                                int N, int K) {
    const unsigned long long ga = (unsigned long long)(size_t)gptr;
    u32x4 g0;
    g0[0] = 1u;                                                  // count=1, user desc
    g0[1] = lds_byte;                                            // lds_addr
    g0[2] = (unsigned)ga;                                        // global_addr[31:0]
    g0[3] = (unsigned)((ga >> 32) & 0x01FFFFFFu) | (2u << 30);   // ga[56:32], type=2
    i32x8 g1;
    g1[0] = (int)(2u << 16);                                     // data_size = 4B
    g1[1] = (int)(((unsigned)N & 0xFFFFu) << 16);                // tensor_dim0[15:0]
    g1[2] = (int)(((unsigned)N >> 16) | (((unsigned)K & 0xFFFFu) << 16)); // d0 hi, d1 lo
    g1[3] = (int)(((unsigned)K >> 16) | (128u << 16));           // d1 hi, tile_dim0=128
    g1[4] = (int)32u;                                            // tile_dim1=32, tile_dim2=0
    g1[5] = (int)(unsigned)N;                                    // tensor_dim0_stride
    g1[6] = 0;                                                   // stride hi, d1_stride lo
    g1[7] = 0;
    const i32x4 z4 = {0, 0, 0, 0};
    const i32x8 z8 = {0, 0, 0, 0, 0, 0, 0, 0};
    __builtin_amdgcn_tensor_load_to_lds(g0, g1, z4, z4, z8, 0);
}

// ---------------------------------------------------------------------------
// Generic GEMM: C[M,N] = epilogue(A[M,K] * B[K,N])
// EPI: 0 none | 1 +bias, exact GELU | 2 +residual | 3 +bias +residual
//      4 head-split store: C laid out [b][head][l][64] (head = col%12)
// Block 256 thr (8 waves), tile 128x128, K-step 32, LDS ping-pong + TDM.
// ---------------------------------------------------------------------------
template<int EPI>
__global__ __launch_bounds__(256)
void gemm_wmma(const float* __restrict__ A, const float* __restrict__ B,
               float* __restrict__ C, const float* __restrict__ bias,
               const float* __restrict__ resid, int M, int N, int K)
{
    __shared__ __align__(16) unsigned int Af[2][8][256];  // A frag images
    __shared__ __align__(16) unsigned int Bf[2][8][256];  // B frag images
    __shared__ __align__(16) float        Rb[2][32][128]; // raw fp32 B tiles (TDM dest)

    const int tid  = threadIdx.x;
    const int lane = tid & 31;
    const int hi   = lane >> 4;
    const int l16  = lane & 15;
    const int wid  = tid >> 5;
    const int wm   = wid & 3;                 // 4 waves over M
    const int wn   = wid >> 2;                // 2 waves over N
    const int row0 = blockIdx.y * 128;
    const int col0 = blockIdx.x * 128;

    // staging assignments
    const int am   = tid >> 1;                // A: row 0..127
    const int ah16 = (tid & 1) << 4;          // A: k-half 0/16
    const int bkp  = (tid >> 4) << 1;         // B: even k row 0,2,..30
    const int bn0  = (tid & 15) << 3;         // B: col 0..120 step 8

    auto stageA = [&](int k0, int buf) {
        const float4* ap = reinterpret_cast<const float4*>(
            A + (size_t)(row0 + am) * K + k0 + ah16);
        float fv[16];
        #pragma unroll
        for (int i = 0; i < 4; ++i) {
            float4 f = ap[i];
            fv[4*i+0] = f.x; fv[4*i+1] = f.y; fv[4*i+2] = f.z; fv[4*i+3] = f.w;
        }
        __builtin_prefetch(A + (size_t)(row0 + am) * K + k0 + 32 + ah16, 0, 0);
        unsigned int* dst = Af[buf][am >> 4];
        #pragma unroll
        for (int j = 0; j < 8; ++j) {
            const int k = ah16 + 2 * j;
            dst[((am & 15) + (khi(k) << 4)) * 8 + kv(k)] = pack2(fv[2*j], fv[2*j+1]);
        }
    };
    // convert + permute the TDM-fetched raw tile into fragment order
    auto stageBperm = [&](int rbuf, int buf) {
        const float4* r0 = reinterpret_cast<const float4*>(&Rb[rbuf][bkp][bn0]);
        const float4* r1 = reinterpret_cast<const float4*>(&Rb[rbuf][bkp + 1][bn0]);
        float lo[8], hg[8];
        #pragma unroll
        for (int i = 0; i < 2; ++i) {
            float4 f = r0[i];
            lo[4*i+0]=f.x; lo[4*i+1]=f.y; lo[4*i+2]=f.z; lo[4*i+3]=f.w;
            float4 g = r1[i];
            hg[4*i+0]=g.x; hg[4*i+1]=g.y; hg[4*i+2]=g.z; hg[4*i+3]=g.w;
        }
        const int hb = khi(bkp), vb = kv(bkp);
        #pragma unroll
        for (int i = 0; i < 8; ++i) {
            const int n = bn0 + i;
            Bf[buf][n >> 4][((n & 15) + (hb << 4)) * 8 + vb] = pack2(lo[i], hg[i]);
        }
    };
    auto issueTDM = [&](int k0, int rbuf) {
        tdm_load_b_tile(B + (size_t)k0 * N + col0,
                        (unsigned)(size_t)(void*)&Rb[rbuf][0][0], N, K);
    };

    v8f acc[2][4];
    #pragma unroll
    for (int mi = 0; mi < 2; ++mi)
        #pragma unroll
        for (int ni = 0; ni < 4; ++ni)
            acc[mi][ni] = v8f{0.f,0.f,0.f,0.f,0.f,0.f,0.f,0.f};

    // ---- prologue: TDM tile 0, stage A0, permute B0, TDM tile 1 ----
    if (wid == 0) issueTDM(0, 0);
    stageA(0, 0);
    if (wid == 0) __builtin_amdgcn_s_wait_tensorcnt(0);
    __syncthreads();                           // Rb[0] ready for all waves
    stageBperm(0, 0);
    if (wid == 0 && 32 < K) issueTDM(32, 1);
    __syncthreads();                           // Af[0], Bf[0] ready

    int cur = 0;
    for (int k0 = 0; k0 < K; k0 += 32) {
        // ---- consume buf `cur` ----
        v16bf af[2], bfr[4];
        #pragma unroll
        for (int mi = 0; mi < 2; ++mi) af[mi]  = load_frag(Af[cur][wm*2 + mi], lane);
        #pragma unroll
        for (int ni = 0; ni < 4; ++ni) bfr[ni] = load_frag(Bf[cur][wn*4 + ni], lane);
        #pragma unroll
        for (int mi = 0; mi < 2; ++mi)
            #pragma unroll
            for (int ni = 0; ni < 4; ++ni)
                acc[mi][ni] = __builtin_amdgcn_wmma_f32_16x16x32_bf16(
                    false, af[mi], false, bfr[ni], (short)0, acc[mi][ni],
                    false, false);
        // ---- prepare buf cur^1 ----
        if (k0 + 32 < K) {
            stageA(k0 + 32, cur ^ 1);
            if (wid == 0) __builtin_amdgcn_s_wait_tensorcnt(0);
            __syncthreads();                   // Rb[cur^1] ready
            stageBperm(cur ^ 1, cur ^ 1);
            if (wid == 0 && k0 + 64 < K) issueTDM(k0 + 64, cur); // Rb[cur] is free
            __syncthreads();                   // Af/Bf[cur^1] ready
        }
        cur ^= 1;
    }

    // ---- epilogue: D layout row = v + 8*hi, col = lane&15 ----
    #pragma unroll
    for (int mi = 0; mi < 2; ++mi) {
        #pragma unroll
        for (int ni = 0; ni < 4; ++ni) {
            const int gr0 = row0 + wm * 32 + mi * 16 + hi * 8;
            const int gc  = col0 + wn * 64 + ni * 16 + l16;
            float bv = 0.f;
            if constexpr (EPI == 1 || EPI == 3) bv = bias[gc];
            int hsp = 0, dsp = 0;
            if constexpr (EPI == 4) { hsp = gc % 12; dsp = gc / 12; }
            #pragma unroll
            for (int v = 0; v < 8; ++v) {
                const int gr = gr0 + v;
                float val = acc[mi][ni][v] + bv;
                if constexpr (EPI == 1)
                    val = 0.5f * val * (1.f + erff(val * 0.70710678118654752f));
                if constexpr (EPI == 2 || EPI == 3)
                    val += resid[(size_t)gr * N + gc];
                if constexpr (EPI == 4) {
                    // row = b*1024 + l ; col = dd*12 + h  ->  [b][h][l][dd]
                    const int bb = gr >> 10, l = gr & 1023;
                    C[((((size_t)bb * 12 + hsp) << 10) + l) * 64 + dsp] = val;
                } else {
                    C[(size_t)gr * N + gc] = val;
                }
            }
        }
    }
}

// ---------------------------------------------------------------------------
// Flash-style attention, faithful to reference:
//   scores = Q.K^T / 8, softmax over m, ctx = P.K  (K, not V).
// Inputs Qh/Kh are head-split [b][head][l][64] (written by gemm_wmma<4>).
// Output Ctx is model layout (b,l, dd*12+head). One wave per block, one
// 16-row l-tile per (batch,head), m in steps of 32. All LDS operands are
// kept in WMMA fragment order -> fragment loads are 2 x ds_load_b128.
// ---------------------------------------------------------------------------
__global__ __launch_bounds__(32)
void attn_wmma(const float* __restrict__ Qh, const float* __restrict__ Kh,
               float* __restrict__ Ctx, int L, int H, int NH)
{
    __shared__ __align__(16) unsigned int Qf[2][256];     // A: Q, k-step = d/32
    __shared__ __align__(16) unsigned int KS[2][2][256];  // B scores: [kstep][m-sub]
    __shared__ __align__(16) unsigned int KC[4][256];     // B ctx: k=m, [d-sub]
    __shared__ __align__(16) unsigned int Pf[256];        // A: probs 16x32
    __shared__ float Sf[16][33];
    __shared__ float lm[16], ls[16], rs[16];

    const int lane = threadIdx.x;
    const int hi   = lane >> 4;
    const int l16  = lane & 15;
    const int b    = blockIdx.z;
    const int head = blockIdx.y;
    const int l0   = blockIdx.x * 16;

    const float* Qb = Qh + (((size_t)b * NH + head) * L + l0) * 64;
    const float* Kb = Kh + (((size_t)b * NH + head) * L) * 64;

    // ---- stage Q tile (16x64) into fragment order, once ----
    {
        const int r = l16, dh = hi;            // row, d-half (0..31 / 32..63)
        const float4* qp = reinterpret_cast<const float4*>(Qb + (size_t)r * 64 + dh * 32);
        float q[32];
        #pragma unroll
        for (int i = 0; i < 8; ++i) {
            float4 f = qp[i];
            q[4*i+0]=f.x; q[4*i+1]=f.y; q[4*i+2]=f.z; q[4*i+3]=f.w;
        }
        #pragma unroll
        for (int j = 0; j < 16; ++j) {
            const int k = 2 * j;
            Qf[dh][(r + (khi(k) << 4)) * 8 + kv(k)] = pack2(q[2*j], q[2*j+1]);
        }
    }
    if (lane < 16) { lm[lane] = -3.0e38f; ls[lane] = 0.f; }

    v8f acc[4];
    #pragma unroll
    for (int t = 0; t < 4; ++t) acc[t] = v8f{0.f,0.f,0.f,0.f,0.f,0.f,0.f,0.f};
    __syncthreads();

    for (int m0 = 0; m0 < L; m0 += 32) {
        // ---- stage K tile (32x64): two m-rows per lane, one d-half ----
        {
            const int r0 = (l16 << 1), r1 = r0 + 1, dh = hi;
            const float4* k0p = reinterpret_cast<const float4*>(
                Kb + (size_t)(m0 + r0) * 64 + dh * 32);
            const float4* k1p = reinterpret_cast<const float4*>(
                Kb + (size_t)(m0 + r1) * 64 + dh * 32);
            float a[32], c[32];
            #pragma unroll
            for (int i = 0; i < 8; ++i) {
                float4 f = k0p[i];
                a[4*i+0]=f.x; a[4*i+1]=f.y; a[4*i+2]=f.z; a[4*i+3]=f.w;
                float4 g = k1p[i];
                c[4*i+0]=g.x; c[4*i+1]=g.y; c[4*i+2]=g.z; c[4*i+3]=g.w;
            }
            // scores-B image: k-axis = d (pairs along d within each row)
            #pragma unroll
            for (int j = 0; j < 16; ++j) {
                const int k = 2 * j;
                const int slot = (khi(k) << 4) * 8 + kv(k);
                KS[dh][r0 >> 4][((r0 & 15)) * 8 + slot] = pack2(a[2*j], a[2*j+1]);
                KS[dh][r1 >> 4][((r1 & 15)) * 8 + slot] = pack2(c[2*j], c[2*j+1]);
            }
            // ctx-B image: k-axis = m (pairs across the two rows)
            const int hm = khi(r0), vm = kv(r0);
            #pragma unroll
            for (int i = 0; i < 32; ++i) {
                const int dd = dh * 32 + i;
                KC[dd >> 4][((dd & 15) + (hm << 4)) * 8 + vm] = pack2(a[i], c[i]);
            }
        }
        __syncthreads();

        // ---- scores: 2 m-subtiles x 2 k-steps of WMMA ----
        v16bf aq[2];
        #pragma unroll
        for (int s = 0; s < 2; ++s) aq[s] = load_frag(Qf[s], lane);
        #pragma unroll
        for (int sh = 0; sh < 2; ++sh) {
            v8f sacc = v8f{0.f,0.f,0.f,0.f,0.f,0.f,0.f,0.f};
            #pragma unroll
            for (int s = 0; s < 2; ++s)
                sacc = __builtin_amdgcn_wmma_f32_16x16x32_bf16(
                    false, aq[s], false, load_frag(KS[s][sh], lane),
                    (short)0, sacc, false, false);
            #pragma unroll
            for (int v = 0; v < 8; ++v)
                Sf[v + 8 * hi][sh * 16 + l16] = sacc[v] * 0.125f; // 1/sqrt(64)
        }
        __syncthreads();

        // ---- online softmax over 32 columns; write probs in frag order ----
        if (lane < 16) {
            const int r = lane;
            float mx = lm[r];
            for (int j = 0; j < 32; ++j) mx = fmaxf(mx, Sf[r][j]);
            const float sc = expf(lm[r] - mx);
            float sum = ls[r] * sc;
            float p[32];
            for (int j = 0; j < 32; ++j) {
                p[j] = expf(Sf[r][j] - mx);
                sum += p[j];
            }
            #pragma unroll
            for (int j = 0; j < 16; ++j) {
                const int k = 2 * j;
                Pf[(r + (khi(k) << 4)) * 8 + kv(k)] = pack2(p[2*j], p[2*j+1]);
            }
            lm[r] = mx; ls[r] = sum; rs[r] = sc;
        }
        __syncthreads();

        // ---- rescale running ctx, then ctx += P(16x32) * K(32x64) ----
        const v16bf pf = load_frag(Pf, lane);
        #pragma unroll
        for (int t = 0; t < 4; ++t) {
            #pragma unroll
            for (int v = 0; v < 8; ++v) acc[t][v] *= rs[v + 8 * hi];
            acc[t] = __builtin_amdgcn_wmma_f32_16x16x32_bf16(
                false, pf, false, load_frag(KC[t], lane),
                (short)0, acc[t], false, false);
        }
        __syncthreads();
    }

    // ---- finalize: divide by row sum; store model layout (dd*NH + head) ----
    #pragma unroll
    for (int t = 0; t < 4; ++t) {
        #pragma unroll
        for (int v = 0; v < 8; ++v) {
            const int r  = v + 8 * hi;
            const int dd = t * 16 + l16;
            Ctx[((size_t)b * L + l0 + r) * H + dd * NH + head] = acc[t][v] / ls[r];
        }
    }
}

// ---------------------------------------------------------------------------
// Row LayerNorm: one block per row of H=768.
// ---------------------------------------------------------------------------
__global__ __launch_bounds__(256)
void layernorm_kernel(const float* __restrict__ in, const float* __restrict__ g,
                      const float* __restrict__ b, float* __restrict__ out, int H)
{
    __shared__ float s1[256], s2[256];
    const int t = threadIdx.x;
    const float* p = in + (size_t)blockIdx.x * H;
    float sum = 0.f, sq = 0.f;
    for (int i = t; i < H; i += 256) { const float v = p[i]; sum += v; sq += v * v; }
    s1[t] = sum; s2[t] = sq;
    __syncthreads();
    for (int s = 128; s > 0; s >>= 1) {
        if (t < s) { s1[t] += s1[t + s]; s2[t] += s2[t + s]; }
        __syncthreads();
    }
    const float mu  = s1[0] / (float)H;
    const float var = s2[0] / (float)H - mu * mu;
    const float inv = rsqrtf(var + 1e-5f);
    for (int i = t; i < H; i += 256)
        out[(size_t)blockIdx.x * H + i] = (p[i] - mu) * inv * g[i] + b[i];
}

// ---------------------------------------------------------------------------
extern "C" void kernel_launch(void* const* d_in, const int* in_sizes, int n_in,
                              void* d_out, int out_size, void* d_ws, size_t ws_size,
                              hipStream_t stream)
{
    (void)in_sizes; (void)n_in; (void)out_size; (void)ws_size;
    const float* x   = (const float*)d_in[0];
    const float* Wq  = (const float*)d_in[1];
    const float* Wk  = (const float*)d_in[2];
    /* d_in[3] = Wv : dead in the reference -> skipped */
    const float* Wo  = (const float*)d_in[4];
    const float* W1  = (const float*)d_in[5];
    const float* b1  = (const float*)d_in[6];
    const float* W2  = (const float*)d_in[7];
    const float* b2  = (const float*)d_in[8];
    const float* g1  = (const float*)d_in[9];
    const float* be1 = (const float*)d_in[10];
    const float* g2  = (const float*)d_in[11];
    const float* be2 = (const float*)d_in[12];

    constexpr int Hh = 768, MLPN = 3072, Lh = 1024, Bb = 8, NHh = 12;
    const int Mrows = Bb * Lh;               // 8192
    const size_t SZ = (size_t)Mrows * Hh;    // 6.29M floats

    float* ws = (float*)d_ws;                // needs ~176 MB of scratch
    float* Qh = ws;                          // [SZ]  head-split [b][h][l][64]
    float* Kh = ws + SZ;                     // [SZ]  head-split
    float* Cx = ws + 2 * SZ;                 // [SZ]  ctx, model layout
    float* H1 = ws + 3 * SZ;                 // [4*SZ] MLP hidden
    float* Y1 = Qh;                          // reuse: Q dead after attention
    float* X2 = Kh;                          // reuse: K dead after attention
    float* Y2 = Cx;                          // reuse: ctx dead after Wo GEMM

    const dim3 blk(256);
    const dim3 g768(Hh / 128, Mrows / 128);
    const dim3 g3072(MLPN / 128, Mrows / 128);

    // Q,K projections, stored head-split for contiguous attention tiles
    gemm_wmma<4><<<g768, blk, 0, stream>>>(x, Wq, Qh, nullptr, nullptr, Mrows, Hh, Hh);
    gemm_wmma<4><<<g768, blk, 0, stream>>>(x, Wk, Kh, nullptr, nullptr, Mrows, Hh, Hh);
    // ctx = softmax(Q.K^T/8).K  per head
    attn_wmma<<<dim3(Lh / 16, NHh, Bb), dim3(32), 0, stream>>>(Qh, Kh, Cx, Lh, Hh, NHh);
    // y1 = ctx@Wo + x ; x2 = LN1(y1)
    gemm_wmma<2><<<g768, blk, 0, stream>>>(Cx, Wo, Y1, nullptr, x, Mrows, Hh, Hh);
    layernorm_kernel<<<Mrows, 256, 0, stream>>>(Y1, g1, be1, X2, Hh);
    // h1 = gelu(x2@W1 + b1) ; y2 = h1@W2 + b2 + x2 ; out = LN2(y2)
    gemm_wmma<1><<<g3072, blk, 0, stream>>>(X2, W1, H1, b1, nullptr, Mrows, MLPN, Hh);
    gemm_wmma<3><<<g768, blk, 0, stream>>>(H1, W2, Y2, b2, X2, Mrows, Hh, MLPN);
    layernorm_kernel<<<Mrows, 256, 0, stream>>>(Y2, g2, be2, (float*)d_out, Hh);
}
